// ODEBlock_38354057954067
// MI455X (gfx1250) — matrix operand order repaired and checked by reading, not compiled, not run on hardware
//
#include <hip/hip_runtime.h>

typedef __attribute__((ext_vector_type(16))) _Float16 v16h;
typedef __attribute__((ext_vector_type(8)))  float    v8f;
typedef unsigned int v4u __attribute__((ext_vector_type(4)));
typedef int          v4i __attribute__((ext_vector_type(4)));
typedef int          v8i __attribute__((ext_vector_type(8)));

#define TS   6553600   // 128*32*40*40 floats
#define TS4  1638400   // TS/4
#define HW   1600      // 40*40
#define NPOS 204800    // 128*40*40 (GEMM M)
#define CONV_BLOCKS 1600   // NPOS / (16 pos/wave * 8 waves/block)

// ---------------------------------------------------------------------------
// Weight prepack: W (OIHW f32, 32x32x3x3) -> f16 B-fragments.
// Bp[((rs*2+nt)*32 + lane)*16 + e]; B lane holds column n = lane&15,
// K = (lane>>4)*8 + (e&7) + (e>>3)*16 within the chunk; chunk rs selects tap (r,s), c = K.
// ---------------------------------------------------------------------------
__global__ void prep_w_kernel(const float* __restrict__ W, _Float16* __restrict__ Bp) {
    int idx = blockIdx.x * blockDim.x + threadIdx.x;
    if (idx >= 9216) return;
    int e  = idx & 15;
    int l  = (idx >> 4) & 31;
    int nt = (idx >> 9) & 1;
    int rs = idx >> 10;            // 0..8
    int r = rs / 3, s = rs - 3 * (rs / 3);
    int k = nt * 16 + (l & 15);    // output channel (GEMM N)
    int c = (l >> 4) * 8 + (e & 7) + (e >> 3) * 16;     // input channel  (GEMM K within chunk)
    float w = W[((k * 32 + c) * 3 + r) * 3 + s];
    Bp[idx] = (_Float16)w;
}

// ---------------------------------------------------------------------------
// Stage packed weights (18432 B) into LDS via the Tensor Data Mover.
// One wave issues a D# describing a 1-D tile of 2304 x 8B elements; completion
// is tracked with TENSORcnt (s_wait_tensorcnt 0) before the workgroup barrier.
// Fallback: cooperative b128 copy loop if the TDM builtin is unavailable.
// ---------------------------------------------------------------------------
__device__ __forceinline__ void stage_weights(const _Float16* __restrict__ Bp,
                                              _Float16* Bs) {
#if __has_builtin(__builtin_amdgcn_tensor_load_to_lds)
    if ((threadIdx.x >> 5) == 0) {          // wave 0 issues the DMA
        unsigned long long ga = (unsigned long long)Bp;
        unsigned int       la = (unsigned int)(unsigned long long)Bs;
        // D# group 0: count=1 | lds_addr | global_addr[56:0] | type=2
        v4u g0;
        g0.x = 1u;
        g0.y = la;
        g0.z = (unsigned int)ga;
        g0.w = (unsigned int)((ga >> 32) & 0x01ffffffu) | (2u << 30);
        // D# group 1: data_size=3 (8B), tensor_dim0=tile_dim0=2304, dim1=1
        v8i g1;
        g1[0] = (3 << 16);           // [17:16] data_size, mask=0, no pad/iterate
        g1[1] = (int)(2304u << 16);  // tensor_dim0[15:0] @ bits[63:48]
        g1[2] = (int)(1u << 16);     // tensor_dim0[31:16]=0 | tensor_dim1[15:0]=1
        g1[3] = (int)(2304u << 16);  // tensor_dim1[31:16]=0 | tile_dim0=2304
        g1[4] = 1;                   // tile_dim1=1, tile_dim2=0
        g1[5] = 2304;                // tensor_dim0_stride[31:0]
        g1[6] = 0;
        g1[7] = 0;
        v4i gz = {0, 0, 0, 0};
#if __clang_major__ >= 23
        v8i gz8 = {0, 0, 0, 0, 0, 0, 0, 0};
        __builtin_amdgcn_tensor_load_to_lds(g0, g1, gz, gz, gz8, 0);
#else
        __builtin_amdgcn_tensor_load_to_lds(g0, g1, gz, gz, 0);
#endif
        __builtin_amdgcn_s_wait_tensorcnt(0);
    }
    __syncthreads();
#else
    for (int i = threadIdx.x; i < 1152; i += 256)
        ((float4*)Bs)[i] = ((const float4*)Bp)[i];
    __syncthreads();
#endif
}

// ---------------------------------------------------------------------------
// Implicit-GEMM conv via v_wmma_f32_16x16x32_f16 on NHWC f32 activations.
// One wave -> 16 spatial positions x 32 output channels. 9 chunks of K=32.
// NHWC: element (position p, channel c) at in[p*32 + c]; tap p2 = p + dy*40+dx.
// A-fragment lane layout = two contiguous 8-float runs (ISA 16-bit A 16x32).
// B fragments staged in LDS once per block (reused by 8 waves x 9 chunks).
// ---------------------------------------------------------------------------
__global__ __launch_bounds__(256) void conv_wmma_kernel(
    const float* __restrict__ in, const _Float16* __restrict__ Bp,
    const float* __restrict__ bias, float* __restrict__ out, int relu)
{
    __shared__ alignas(16) _Float16 Bs[9216];
    stage_weights(Bp, Bs);

    int lane = threadIdx.x & 31;
    int wave = threadIdx.x >> 5;
    int tile = blockIdx.x * 8 + wave;
    int p0   = tile << 4;                 // base GEMM row (spatial position)
    int m    = lane & 15;
    int half = lane >> 4;
    int p    = p0 + m;
    int nimg = p / HW;
    int rem  = p - nimg * HW;
    int hh   = rem / 40;
    int wwp  = rem - hh * 40;

    v8f acc0 = {};
    v8f acc1 = {};

#pragma unroll
    for (int rs = 0; rs < 9; ++rs) {
        int dy = rs / 3 - 1;
        int dx = rs - 3 * (rs / 3) - 1;
        int h2 = hh + dy, w2 = wwp + dx;
        bool valid = (h2 >= 0) && (h2 < 40) && (w2 >= 0) && (w2 < 40);
        const float* src = in + (size_t)(p + dy * 40 + dx) * 32;

        if (rs < 8) {  // prefetch next tap row (global_prefetch_b8)
            int dyn = (rs + 1) / 3 - 1;
            int dxn = (rs + 1) - 3 * ((rs + 1) / 3) - 1;
            __builtin_prefetch(in + (size_t)(p + dyn * 40 + dxn) * 32, 0, 0);
        }

        // Lane's 16 A-halves = channels [half*8 .. half*8+7] and
        // [16+half*8 .. 16+half*8+7], in element order (4x coalesced b128).
        float4 r0a, r0b, r1a, r1b;
        if (valid) {
            r0a = *(const float4*)(src + half * 8);
            r0b = *(const float4*)(src + half * 8 + 4);
            r1a = *(const float4*)(src + 16 + half * 8);
            r1b = *(const float4*)(src + 16 + half * 8 + 4);
        } else {
            r0a = r0b = r1a = r1b = make_float4(0.f, 0.f, 0.f, 0.f);
        }
        v16h a;
        a[0]  = (_Float16)r0a.x; a[1]  = (_Float16)r0a.y;
        a[2]  = (_Float16)r0a.z; a[3]  = (_Float16)r0a.w;
        a[4]  = (_Float16)r0b.x; a[5]  = (_Float16)r0b.y;
        a[6]  = (_Float16)r0b.z; a[7]  = (_Float16)r0b.w;
        a[8]  = (_Float16)r1a.x; a[9]  = (_Float16)r1a.y;
        a[10] = (_Float16)r1a.z; a[11] = (_Float16)r1a.w;
        a[12] = (_Float16)r1b.x; a[13] = (_Float16)r1b.y;
        a[14] = (_Float16)r1b.z; a[15] = (_Float16)r1b.w;

        // B fragments from LDS: 16 contiguous halves per lane (2x ds_load_b128)
        v16h b0 = *(const v16h*)(Bs + ((rs * 2 + 0) * 32 + lane) * 16);
        v16h b1 = *(const v16h*)(Bs + ((rs * 2 + 1) * 32 + lane) * 16);

        acc0 = __builtin_amdgcn_wmma_f32_16x16x32_f16(false, a, false, b0,
                                                      (short)0, acc0, false, false);
        acc1 = __builtin_amdgcn_wmma_f32_16x16x32_f16(false, a, false, b1,
                                                      (short)0, acc1, false, false);
    }

    // C/D layout: lane -> column n = lane&15 (+16 for acc1), VGPR r -> row r+8*half
    // NHWC store: for fixed row, lanes 0..15 write contiguous 64B channel runs.
    int col = lane & 15;
    float bias0 = bias[col];
    float bias1 = bias[col + 16];
#pragma unroll
    for (int r = 0; r < 8; ++r) {
        int pp  = p0 + r + half * 8;
        float v0 = acc0[r] + bias0;
        float v1 = acc1[r] + bias1;
        if (relu) { v0 = fmaxf(v0, 0.0f); v1 = fmaxf(v1, 0.0f); }
        out[(size_t)pp * 32 + col]      = v0;
        out[(size_t)pp * 32 + col + 16] = v1;
    }
}

// ---------------------------------------------------------------------------
// Elementwise stage combine: out = y + h * sum(ci * ki), float4 streaming.
// ---------------------------------------------------------------------------
__device__ inline float4 fma4(float c, float4 k, float4 a) {
    a.x += c * k.x; a.y += c * k.y; a.z += c * k.z; a.w += c * k.w; return a;
}

__global__ __launch_bounds__(256) void combine_kernel(
    const float4* __restrict__ y,
    const float4* k1, const float4* k2, const float4* k3,
    const float4* k4, const float4* k5, const float4* k6,
    float c1, float c2, float c3, float c4, float c5, float c6,
    const float* __restrict__ ctrl, float4* __restrict__ out)
{
    int i = blockIdx.x * blockDim.x + threadIdx.x;
    if (i >= TS4) return;
    float h = ctrl[4];
    float4 acc = make_float4(0.f, 0.f, 0.f, 0.f);
    acc = fma4(c1, k1[i], acc);
    if (k2) acc = fma4(c2, k2[i], acc);
    if (k3) acc = fma4(c3, k3[i], acc);
    if (k4) acc = fma4(c4, k4[i], acc);
    if (k5) acc = fma4(c5, k5[i], acc);
    if (k6) acc = fma4(c6, k6[i], acc);
    float4 yy = y[i];
    yy.x += h * acc.x; yy.y += h * acc.y; yy.z += h * acc.z; yy.w += h * acc.w;
    out[i] = yy;
}

// ---------------------------------------------------------------------------
// Error norm pass 1: per-block partial sums of (err/tol)^2 (deterministic).
// ---------------------------------------------------------------------------
__global__ __launch_bounds__(256) void err_partial_kernel(
    const float* __restrict__ y,  const float* __restrict__ y5,
    const float* __restrict__ k1, const float* __restrict__ k3,
    const float* __restrict__ k4, const float* __restrict__ k5,
    const float* __restrict__ k6, const float* __restrict__ k7,
    const float* __restrict__ ctrl, float* __restrict__ partials)
{
    const float E0 = 35.f/384.f  - 5179.f/57600.f;
    const float E2 = 500.f/1113.f - 7571.f/16695.f;
    const float E3 = 125.f/192.f  - 393.f/640.f;
    const float E4 = -2187.f/6784.f + 92097.f/339200.f;
    const float E5 = 11.f/84.f - 187.f/2100.f;
    const float E6 = -1.f/40.f;
    float h = ctrl[4];
    int stride = gridDim.x * blockDim.x;
    float acc = 0.f;
    for (int i = blockIdx.x * blockDim.x + threadIdx.x; i < TS; i += stride) {
        float err = h * (E0 * k1[i] + E2 * k3[i] + E3 * k4[i]
                       + E4 * k5[i] + E5 * k6[i] + E6 * k7[i]);
        float tol = 1e-4f + 1e-3f * fmaxf(fabsf(y[i]), fabsf(y5[i]));
        float q = err / tol;
        acc += q * q;
    }
    __shared__ float s[256];
    s[threadIdx.x] = acc;
    __syncthreads();
    for (int off = 128; off > 0; off >>= 1) {
        if (threadIdx.x < off) s[threadIdx.x] += s[threadIdx.x + off];
        __syncthreads();
    }
    if (threadIdx.x == 0) partials[blockIdx.x] = s[0];
}

// ---------------------------------------------------------------------------
// Error norm pass 2 + dopri5 controller (single block, deterministic).
// ctrl: [0]=t [1]=dt [2]=done [3]=accept [4]=h
// ---------------------------------------------------------------------------
__global__ __launch_bounds__(256) void finalize_kernel(
    const float* __restrict__ partials, float* __restrict__ ctrl)
{
    __shared__ float s[256];
    int tid = threadIdx.x;
    s[tid] = partials[tid] + partials[tid + 256] + partials[tid + 512] + partials[tid + 768];
    __syncthreads();
    for (int off = 128; off > 0; off >>= 1) {
        if (tid < off) s[tid] += s[tid + off];
        __syncthreads();
    }
    if (tid == 0) {
        float en   = sqrtf(s[0] / (float)TS);
        float t    = ctrl[0], dt = ctrl[1], done = ctrl[2], h = ctrl[4];
        bool  acc  = (en <= 1.0f) && (done == 0.0f);
        float t_n  = acc ? (t + h) : t;
        float fac  = 0.9f * powf(en + 1e-10f, -0.2f);
        fac        = fminf(fmaxf(fac, 0.2f), 10.0f);
        float dt_n = (done != 0.0f) ? dt : dt * fac;
        ctrl[0] = t_n;
        ctrl[1] = dt_n;
        ctrl[2] = (t_n >= 1.0f - 1e-6f) ? 1.0f : 0.0f;
        ctrl[3] = acc ? 1.0f : 0.0f;
    }
}

__global__ __launch_bounds__(256) void begin_step_kernel(float* ctrl) {
    if (threadIdx.x == 0) ctrl[4] = fminf(ctrl[1], 1.0f - ctrl[0]);
}

__global__ __launch_bounds__(256) void y_update_kernel(
    float4* __restrict__ y, const float4* __restrict__ y5, const float* __restrict__ ctrl)
{
    if (ctrl[3] == 0.0f) return;   // uniform: rejected step or already done
    int i = blockIdx.x * blockDim.x + threadIdx.x;
    if (i < TS4) y[i] = y5[i];
}

// NCHW f32 input -> NHWC f32 working layout; also initializes controller.
__global__ __launch_bounds__(256) void init_kernel(
    const float* __restrict__ x, float* __restrict__ y, float* __restrict__ ctrl)
{
    int i = blockIdx.x * blockDim.x + threadIdx.x;   // NHWC flat index
    if (i < TS) {
        int c = i & 31;
        int p = i >> 5;                // n*1600 + hw
        int n = p / HW;
        int hw = p - n * HW;
        y[i] = x[((size_t)n * 32 + c) * HW + hw];
    }
    if (i == 0) { ctrl[0] = 0.0f; ctrl[1] = 0.05f; ctrl[2] = 0.0f; ctrl[3] = 0.0f; ctrl[4] = 0.05f; }
}

// NHWC working layout -> NCHW f32 output.
__global__ __launch_bounds__(256) void copy_out_kernel(
    const float* __restrict__ y, float* __restrict__ dst)
{
    int j = blockIdx.x * blockDim.x + threadIdx.x;   // NCHW flat index
    if (j >= TS) return;
    int n   = j / (32 * HW);
    int rem = j - n * 32 * HW;
    int c   = rem / HW;
    int hw  = rem - c * HW;
    dst[j] = y[((size_t)n * HW + hw) * 32 + c];
}

// ---------------------------------------------------------------------------
extern "C" void kernel_launch(void* const* d_in, const int* in_sizes, int n_in,
                              void* d_out, int out_size, void* d_ws, size_t ws_size,
                              hipStream_t stream) {
    const float* x  = (const float*)d_in[0];
    const float* W1 = (const float*)d_in[1];
    const float* b1 = (const float*)d_in[2];
    const float* W2 = (const float*)d_in[3];
    const float* b2 = (const float*)d_in[4];

    float* y    = (float*)d_ws;
    float* ytmp = y + (size_t)1 * TS;   // also holds y5
    float* z    = y + (size_t)2 * TS;   // conv1 intermediate
    float* k1   = y + (size_t)3 * TS;
    float* k2   = y + (size_t)4 * TS;   // reused as k7
    float* k3   = y + (size_t)5 * TS;
    float* k4   = y + (size_t)6 * TS;
    float* k5   = y + (size_t)7 * TS;
    float* k6   = y + (size_t)8 * TS;
    _Float16* Bp1 = (_Float16*)(y + (size_t)9 * TS);
    _Float16* Bp2 = Bp1 + 9216;
    float* partials = (float*)(Bp2 + 9216);
    float* ctrl     = partials + 1024;

    init_kernel<<<TS / 256, 256, 0, stream>>>(x, y, ctrl);
    prep_w_kernel<<<36, 256, 0, stream>>>(W1, Bp1);
    prep_w_kernel<<<36, 256, 0, stream>>>(W2, Bp2);

    auto f = [&](const float* in, float* out) {  // out = conv2(relu(conv1(in)))
        conv_wmma_kernel<<<CONV_BLOCKS, 256, 0, stream>>>(in, Bp1, b1, z, 1);
        conv_wmma_kernel<<<CONV_BLOCKS, 256, 0, stream>>>(z, Bp2, b2, out, 0);
    };
    auto combine = [&](const float* a1, const float* a2, const float* a3,
                       const float* a4, const float* a5, const float* a6,
                       float c1, float c2, float c3, float c4, float c5, float c6,
                       float* out) {
        combine_kernel<<<TS4 / 256, 256, 0, stream>>>(
            (const float4*)y, (const float4*)a1, (const float4*)a2, (const float4*)a3,
            (const float4*)a4, (const float4*)a5, (const float4*)a6,
            c1, c2, c3, c4, c5, c6, ctrl, (float4*)out);
    };

    for (int step = 0; step < 32; ++step) {
        begin_step_kernel<<<1, 32, 0, stream>>>(ctrl);

        f(y, k1);
        combine(k1, 0, 0, 0, 0, 0, 0.2f, 0, 0, 0, 0, 0, ytmp);
        f(ytmp, k2);
        combine(k1, k2, 0, 0, 0, 0, 0.075f, 0.225f, 0, 0, 0, 0, ytmp);
        f(ytmp, k3);
        combine(k1, k2, k3, 0, 0, 0, 44.f/45.f, -56.f/15.f, 32.f/9.f, 0, 0, 0, ytmp);
        f(ytmp, k4);
        combine(k1, k2, k3, k4, 0, 0,
                19372.f/6561.f, -25360.f/2187.f, 64448.f/6561.f, -212.f/729.f, 0, 0, ytmp);
        f(ytmp, k5);
        combine(k1, k2, k3, k4, k5, 0,
                9017.f/3168.f, -355.f/33.f, 46732.f/5247.f, 49.f/176.f, -5103.f/18656.f, 0, ytmp);
        f(ytmp, k6);
        // y5 into ytmp
        combine(k1, 0, k3, k4, k5, k6,
                35.f/384.f, 0, 500.f/1113.f, 125.f/192.f, -2187.f/6784.f, 11.f/84.f, ytmp);
        f(ytmp, k2);  // k7 (FSAL stage) aliases k2's buffer

        err_partial_kernel<<<1024, 256, 0, stream>>>(y, ytmp, k1, k3, k4, k5, k6, k2,
                                                     ctrl, partials);
        finalize_kernel<<<1, 256, 0, stream>>>(partials, ctrl);
        y_update_kernel<<<TS4 / 256, 256, 0, stream>>>((float4*)y, (const float4*)ytmp, ctrl);
    }

    copy_out_kernel<<<TS / 256, 256, 0, stream>>>(y, (float*)d_out);
    (void)in_sizes; (void)n_in; (void)out_size; (void)ws_size;
}
